// GPT2AttentionWithLoRA_37228776521781
// MI455X (gfx1250) — matrix operand order repaired and checked by reading, not compiled, not run on hardware
//
#include <hip/hip_runtime.h>
#include <hip/hip_bf16.h>
#include <math.h>

typedef __attribute__((ext_vector_type(16))) _Float16 v16h;
typedef __attribute__((ext_vector_type(8)))  float    v8f;

#define D_DIM 1024
#define S_DIM 2048
#define B_DIM 4
#define R_DIM 4
#define LORA_SCALING 8.0f            // 32.0 / r
#define ATTN_SCALE   0.03125f        // 1/sqrt(1024)
#define MASK_VAL   (-3.4028234663852886e38f)

// ---------------- WMMA fragment loaders (wave32, 16x16x32 f16) ----------------
// A: 16x32 (MxK) f16, row-major, leading dim lda.
// Lanes 0-15: row M=lane, K = k0+0..15; lanes 16-31: same rows, K = k0+16..31.
static __device__ __forceinline__ v16h frag_a(const _Float16* A, int lda, int m0, int k0, int lane) {
  const int row = m0 + (lane & 15);
  const int kk  = k0 + ((lane >> 4) << 4);
  const _Float16* p = A + (size_t)row * lda + kk;
  v16h f;
#pragma unroll
  for (int t = 0; t < 16; ++t) f[t] = p[t];
  return f;
}

// B: 32x16 (KxN); lanes 0-15 hold K=0..15 for N=lane, lanes 16-31 hold K=16..31.
// NT source: row-major [N, K] (dot-of-rows) -> 16 contiguous halves per lane.
static __device__ __forceinline__ v16h frag_b_nt(const _Float16* Bm, int ldb, int n0, int k0, int lane) {
  const int col = n0 + (lane & 15);
  const int kk  = k0 + ((lane >> 4) << 4);
  const _Float16* p = Bm + (size_t)col * ldb + kk;
  v16h f;
#pragma unroll
  for (int t = 0; t < 16; ++t) f[t] = p[t];
  return f;
}

static __device__ __forceinline__ v8f wmma_f16(v16h a, v16h b, v8f c) {
  return __builtin_amdgcn_wmma_f32_16x16x32_f16(false, a, false, b, (short)0, c, false, false);
}

// ---------------- helper kernels ----------------
__global__ void k_f32_to_f16(const float* __restrict__ src, _Float16* __restrict__ dst, int n) {
  int i = blockIdx.x * blockDim.x + threadIdx.x;
  if (i < n) dst[i] = (_Float16)src[i];
}

// dst[n*K + k] = (f16) src[k*N + n]   (transpose-convert, for Wo)
__global__ void k_transpose_f16(const float* __restrict__ src, _Float16* __restrict__ dst,
                                int K, int N) {
  int idx = blockIdx.x * blockDim.x + threadIdx.x;
  if (idx >= K * N) return;
  int k = idx / N, n = idx - k * N;
  dst[(size_t)n * K + k] = (_Float16)src[idx];
}

// xA: [M, R] = x[M, D] @ A[D, R]  (tiny, fp32)
__global__ void k_lora_x(const float* __restrict__ x, const float* __restrict__ A,
                         float* __restrict__ out, int M) {
  int idx = blockIdx.x * blockDim.x + threadIdx.x;
  if (idx >= M * R_DIM) return;
  int m = idx >> 2, r = idx & 3;
  float s = 0.f;
  for (int k = 0; k < D_DIM; ++k) s += x[(size_t)m * D_DIM + k] * A[k * R_DIM + r];
  out[idx] = s;
}

// q/k/v projection: out = x @ W^T (+bias) (+ SCALING*(xA)@loraB), f16 result.
// Wave tile: 32 rows x 64 cols (2 A frags x 4 B frags -> 8 WMMA per K-step).
// transposed==0: out[row*D + col].  transposed!=0: out[b*D*S + col*S + i] ([B,D,S] for attn.v).
__global__ void k_proj(const _Float16* __restrict__ xh, const _Float16* __restrict__ Wh,
                       const float* __restrict__ bias, const float* __restrict__ loraX,
                       const float* __restrict__ loraB, _Float16* __restrict__ outh,
                       int transposed) {
  const int lane = threadIdx.x & 31;
  const int wave = threadIdx.x >> 5;
  const int m0 = blockIdx.x << 5;                // 32 rows per wave tile
  const int n0 = (blockIdx.y * 4 + wave) << 6;   // 64 cols per wave
  v8f acc[2][4];
#pragma unroll
  for (int mi = 0; mi < 2; ++mi)
#pragma unroll
    for (int s = 0; s < 4; ++s) acc[mi][s] = (v8f){0.f,0.f,0.f,0.f,0.f,0.f,0.f,0.f};
  for (int k0 = 0; k0 < D_DIM; k0 += 32) {
    v16h a0 = frag_a(xh, D_DIM, m0, k0, lane);
    v16h a1 = frag_a(xh, D_DIM, m0 + 16, k0, lane);
#pragma unroll
    for (int s = 0; s < 4; ++s) {
      v16h b = frag_b_nt(Wh, D_DIM, n0 + 16 * s, k0, lane);
      acc[0][s] = wmma_f16(a0, b, acc[0][s]);
      acc[1][s] = wmma_f16(a1, b, acc[1][s]);
    }
  }
  const int roff = (lane >> 4) << 3;
#pragma unroll
  for (int s = 0; s < 4; ++s) {
    const int col = n0 + 16 * s + (lane & 15);
    float bv = bias ? bias[col] : 0.f;
    float lb0 = 0.f, lb1 = 0.f, lb2 = 0.f, lb3 = 0.f;
    if (loraX) {
      lb0 = loraB[col];             lb1 = loraB[D_DIM + col];
      lb2 = loraB[2 * D_DIM + col]; lb3 = loraB[3 * D_DIM + col];
    }
#pragma unroll
    for (int mi = 0; mi < 2; ++mi) {
      const int rbase = m0 + 16 * mi + roff;
      const int bidx  = rbase >> 11;             // batch (S=2048 rows each; tiles never span)
      const int ibase = rbase & (S_DIM - 1);
#pragma unroll
      for (int g = 0; g < 8; ++g) {
        int row = rbase + g;
        float v = acc[mi][s][g] + bv;
        if (loraX) {
          const float* lx = loraX + (size_t)row * R_DIM;
          v += LORA_SCALING * (lx[0] * lb0 + lx[1] * lb1 + lx[2] * lb2 + lx[3] * lb3);
        }
        if (transposed)
          outh[(size_t)bidx * D_DIM * S_DIM + (size_t)col * S_DIM + (ibase + g)] = (_Float16)v;
        else
          outh[(size_t)row * D_DIM + col] = (_Float16)v;
      }
    }
  }
}

// scores[b,i,j] = scale * q[b,i,:] . k[b,j,:], causal-masked; fp32 into d_out attn_w region.
__global__ void k_scores(const _Float16* __restrict__ qh, const _Float16* __restrict__ kh,
                         float* __restrict__ scores) {
  const int lane = threadIdx.x & 31;
  const int wave = threadIdx.x >> 5;
  const int b  = blockIdx.z;
  const int m0 = blockIdx.x << 5;
  const int n0 = (blockIdx.y * 4 + wave) << 6;
  float* sc = scores + (size_t)b * S_DIM * S_DIM;
  const int roff = (lane >> 4) << 3;
  if (n0 >= m0 + 32) {  // whole 32x64 wave tile above diagonal (wave-uniform)
#pragma unroll
    for (int s = 0; s < 4; ++s) {
      const int col = n0 + 16 * s + (lane & 15);
#pragma unroll
      for (int mi = 0; mi < 2; ++mi) {
        const int rbase = m0 + 16 * mi + roff;
#pragma unroll
        for (int g = 0; g < 8; ++g) sc[(size_t)(rbase + g) * S_DIM + col] = MASK_VAL;
      }
    }
    return;
  }
  const _Float16* q = qh + (size_t)b * S_DIM * D_DIM;
  const _Float16* k = kh + (size_t)b * S_DIM * D_DIM;
  bool need[2][4];
#pragma unroll
  for (int mi = 0; mi < 2; ++mi)
#pragma unroll
    for (int s = 0; s < 4; ++s) need[mi][s] = (n0 + 16 * s) <= (m0 + 16 * mi);
  v8f acc[2][4];
#pragma unroll
  for (int mi = 0; mi < 2; ++mi)
#pragma unroll
    for (int s = 0; s < 4; ++s) acc[mi][s] = (v8f){0.f,0.f,0.f,0.f,0.f,0.f,0.f,0.f};
  for (int k0 = 0; k0 < D_DIM; k0 += 32) {
    v16h a0 = frag_a(q, D_DIM, m0, k0, lane);
    v16h a1 = frag_a(q, D_DIM, m0 + 16, k0, lane);
#pragma unroll
    for (int s = 0; s < 4; ++s) {
      if (need[0][s] || need[1][s]) {   // wave-uniform predicates; EXEC stays all-1s
        v16h bb = frag_b_nt(k, D_DIM, n0 + 16 * s, k0, lane);
        if (need[0][s]) acc[0][s] = wmma_f16(a0, bb, acc[0][s]);
        if (need[1][s]) acc[1][s] = wmma_f16(a1, bb, acc[1][s]);
      }
    }
  }
#pragma unroll
  for (int s = 0; s < 4; ++s) {
    const int col = n0 + 16 * s + (lane & 15);
#pragma unroll
    for (int mi = 0; mi < 2; ++mi) {
      const int rbase = m0 + 16 * mi + roff;
#pragma unroll
      for (int g = 0; g < 8; ++g) {
        int row = rbase + g;
        float v = need[mi][s] ? (acc[mi][s][g] * ATTN_SCALE) : MASK_VAL;
        if (col > row) v = MASK_VAL;
        sc[(size_t)row * S_DIM + col] = v;
      }
    }
  }
}

// Row softmax over S=2048; fp32 in place + f16 mirror for the attn.v GEMM.
__global__ void k_softmax(float* __restrict__ scores, _Float16* __restrict__ attnh) {
  __shared__ float red[256];
  const int t = threadIdx.x;
  float* sc = scores + (size_t)blockIdx.x * S_DIM;
  _Float16* ah = attnh + (size_t)blockIdx.x * S_DIM;
  float v[8];
#pragma unroll
  for (int i = 0; i < 8; ++i) v[i] = sc[t + i * 256];
  float m = v[0];
#pragma unroll
  for (int i = 1; i < 8; ++i) m = fmaxf(m, v[i]);
  red[t] = m; __syncthreads();
  for (int s = 128; s > 0; s >>= 1) { if (t < s) red[t] = fmaxf(red[t], red[t + s]); __syncthreads(); }
  m = red[0]; __syncthreads();
  float sum = 0.f;
#pragma unroll
  for (int i = 0; i < 8; ++i) { v[i] = expf(v[i] - m); sum += v[i]; }
  red[t] = sum; __syncthreads();
  for (int s = 128; s > 0; s >>= 1) { if (t < s) red[t] += red[t + s]; __syncthreads(); }
  float inv = 1.0f / red[0];
#pragma unroll
  for (int i = 0; i < 8; ++i) {
    float w = v[i] * inv;
    sc[t + i * 256] = w;
    ah[t + i * 256] = (_Float16)w;
  }
}

// pre[b,i,n] = sum_j attn[b,i,j] * v[b,j,n]; v pre-transposed [B, D, S] so B is NT.
// K loop truncated at the causal diagonal (attn rows m0..m0+31 are zero beyond j=m0+31).
__global__ void k_attnv(const _Float16* __restrict__ attnh, const _Float16* __restrict__ vth,
                        _Float16* __restrict__ preh) {
  const int lane = threadIdx.x & 31;
  const int wave = threadIdx.x >> 5;
  const int b  = blockIdx.z;
  const int m0 = blockIdx.x << 5;
  const int n0 = (blockIdx.y * 4 + wave) << 6;
  const _Float16* A  = attnh + (size_t)b * S_DIM * S_DIM;
  const _Float16* VT = vth  + (size_t)b * D_DIM * S_DIM;   // [D, S]
  const int kmax = m0 + 32;                                // 32-aligned
  v8f acc[2][4];
#pragma unroll
  for (int mi = 0; mi < 2; ++mi)
#pragma unroll
    for (int s = 0; s < 4; ++s) acc[mi][s] = (v8f){0.f,0.f,0.f,0.f,0.f,0.f,0.f,0.f};
  for (int k0 = 0; k0 < kmax; k0 += 32) {
    v16h a0 = frag_a(A, S_DIM, m0, k0, lane);
    v16h a1 = frag_a(A, S_DIM, m0 + 16, k0, lane);
#pragma unroll
    for (int s = 0; s < 4; ++s) {
      v16h bb = frag_b_nt(VT, S_DIM, n0 + 16 * s, k0, lane);
      acc[0][s] = wmma_f16(a0, bb, acc[0][s]);
      acc[1][s] = wmma_f16(a1, bb, acc[1][s]);
    }
  }
  const int roff = (lane >> 4) << 3;
  _Float16* out = preh + (size_t)b * S_DIM * D_DIM;
#pragma unroll
  for (int s = 0; s < 4; ++s) {
    const int col = n0 + 16 * s + (lane & 15);
#pragma unroll
    for (int mi = 0; mi < 2; ++mi) {
      const int rbase = m0 + 16 * mi + roff;
#pragma unroll
      for (int g = 0; g < 8; ++g)
        out[(size_t)(rbase + g) * D_DIM + col] = (_Float16)acc[mi][s][g];
    }
  }
}

// out = pre @ Wo + bo, Wo pre-transposed to [out, in] so B is NT; fp32 result.
__global__ void k_outproj(const _Float16* __restrict__ preh, const _Float16* __restrict__ WoTh,
                          const float* __restrict__ bo, float* __restrict__ out) {
  const int lane = threadIdx.x & 31;
  const int wave = threadIdx.x >> 5;
  const int m0 = blockIdx.x << 5;
  const int n0 = (blockIdx.y * 4 + wave) << 6;
  v8f acc[2][4];
#pragma unroll
  for (int mi = 0; mi < 2; ++mi)
#pragma unroll
    for (int s = 0; s < 4; ++s) acc[mi][s] = (v8f){0.f,0.f,0.f,0.f,0.f,0.f,0.f,0.f};
  for (int k0 = 0; k0 < D_DIM; k0 += 32) {
    v16h a0 = frag_a(preh, D_DIM, m0, k0, lane);
    v16h a1 = frag_a(preh, D_DIM, m0 + 16, k0, lane);
#pragma unroll
    for (int s = 0; s < 4; ++s) {
      v16h b = frag_b_nt(WoTh, D_DIM, n0 + 16 * s, k0, lane);
      acc[0][s] = wmma_f16(a0, b, acc[0][s]);
      acc[1][s] = wmma_f16(a1, b, acc[1][s]);
    }
  }
  const int roff = (lane >> 4) << 3;
#pragma unroll
  for (int s = 0; s < 4; ++s) {
    const int col = n0 + 16 * s + (lane & 15);
    float bv = bo[col];
#pragma unroll
    for (int mi = 0; mi < 2; ++mi) {
      const int rbase = m0 + 16 * mi + roff;
#pragma unroll
      for (int g = 0; g < 8; ++g)
        out[(size_t)(rbase + g) * D_DIM + col] = acc[mi][s][g] + bv;
    }
  }
}

// ---------------- launcher ----------------
extern "C" void kernel_launch(void* const* d_in, const int* in_sizes, int n_in,
                              void* d_out, int out_size, void* d_ws, size_t ws_size,
                              hipStream_t stream) {
  (void)in_sizes; (void)n_in; (void)out_size; (void)ws_size;
  const float* x  = (const float*)d_in[0];
  const float* Wq = (const float*)d_in[1];
  const float* bq = (const float*)d_in[2];
  const float* Aq = (const float*)d_in[3];
  const float* Bq = (const float*)d_in[4];
  const float* Wk = (const float*)d_in[5];
  const float* Wv = (const float*)d_in[6];
  const float* bv = (const float*)d_in[7];
  const float* Av = (const float*)d_in[8];
  const float* Bv = (const float*)d_in[9];
  const float* Wo = (const float*)d_in[10];
  const float* bo = (const float*)d_in[11];

  const int M  = B_DIM * S_DIM;              // 8192
  const size_t MD = (size_t)M * D_DIM;       // 8.4M elems
  float* attn_out = (float*)d_out;                           // [B,S,D]
  float* attn_w   = (float*)d_out + MD;                      // [B,S,S]

  char* ws = (char*)d_ws;
  size_t off = 0;
  auto alloc = [&](size_t bytes) -> void* {
    void* p = ws + off;
    off = (off + bytes + 255) & ~(size_t)255;
    return p;
  };
  _Float16* xh    = (_Float16*)alloc(MD * 2);
  _Float16* Wqh   = (_Float16*)alloc((size_t)D_DIM * D_DIM * 2);
  _Float16* Wkh   = (_Float16*)alloc((size_t)D_DIM * D_DIM * 2);
  _Float16* Wvh   = (_Float16*)alloc((size_t)D_DIM * D_DIM * 2);
  _Float16* WoTh  = (_Float16*)alloc((size_t)D_DIM * D_DIM * 2);
  _Float16* qh    = (_Float16*)alloc(MD * 2);
  _Float16* kh    = (_Float16*)alloc(MD * 2);
  _Float16* vth   = (_Float16*)alloc(MD * 2);   // [B, D, S] transposed v
  _Float16* attnh = (_Float16*)alloc((size_t)B_DIM * S_DIM * S_DIM * 2);
  _Float16* preh  = (_Float16*)alloc(MD * 2);
  float*    lxq   = (float*)alloc((size_t)M * R_DIM * 4);
  float*    lxv   = (float*)alloc((size_t)M * R_DIM * 4);

  // 1) f16 conversions (+ Wo transpose-convert)
  k_f32_to_f16<<<(int)(MD / 256), 256, 0, stream>>>(x, xh, (int)MD);
  const int wN = D_DIM * D_DIM;
  k_f32_to_f16<<<wN / 256, 256, 0, stream>>>(Wq, Wqh, wN);
  k_f32_to_f16<<<wN / 256, 256, 0, stream>>>(Wk, Wkh, wN);
  k_f32_to_f16<<<wN / 256, 256, 0, stream>>>(Wv, Wvh, wN);
  k_transpose_f16<<<wN / 256, 256, 0, stream>>>(Wo, WoTh, D_DIM, D_DIM);

  // 2) LoRA down-projections (fp32, tiny)
  k_lora_x<<<(M * R_DIM) / 256, 256, 0, stream>>>(x, Aq, lxq, M);
  k_lora_x<<<(M * R_DIM) / 256, 256, 0, stream>>>(x, Av, lxv, M);

  // 3) q/k/v projections (WMMA, 32x64 wave tiles); v stored transposed
  dim3 gProj(M / 32, D_DIM / 256);
  k_proj<<<gProj, 128, 0, stream>>>(xh, Wqh, bq, lxq, Bq, qh, 0);
  k_proj<<<gProj, 128, 0, stream>>>(xh, Wkh, nullptr, nullptr, nullptr, kh, 0);
  k_proj<<<gProj, 128, 0, stream>>>(xh, Wvh, bv, lxv, Bv, vth, 1);

  // 4) causal scores (WMMA) -> fp32 attn_w region
  dim3 gSc(S_DIM / 32, S_DIM / 256, B_DIM);
  k_scores<<<gSc, 128, 0, stream>>>(qh, kh, attn_w);

  // 5) softmax (fp32 in place + f16 mirror)
  k_softmax<<<M, 256, 0, stream>>>(attn_w, attnh);

  // 6) attn @ v (WMMA, NT via transposed v)
  dim3 gAv(S_DIM / 32, D_DIM / 256, B_DIM);
  k_attnv<<<gAv, 128, 0, stream>>>(attnh, vth, preh);

  // 7) output projection + bias (WMMA, NT via transposed Wo) -> fp32 attn_out
  k_outproj<<<gProj, 128, 0, stream>>>(preh, WoTh, bo, attn_out);
}
